// EngramMemory_70677981823278
// MI455X (gfx1250) — matrix operand order repaired and checked by reading, not compiled, not run on hardware
//
#include <hip/hip_runtime.h>

#define DEV __device__ __forceinline__

typedef __attribute__((ext_vector_type(16))) __bf16 v16bf;
typedef __attribute__((ext_vector_type(8)))  float  v8f;

static constexpr int Bsz  = 8;
static constexpr int Ssz  = 2048;
static constexpr int Dsz  = 64;
static constexpr int NSEG = 24;     // 3 ngrams * 8 heads
static constexpr int KTOT = 1536;   // NSEG * 64
static constexpr int TM   = 16;     // rows per block tile
static constexpr int MSTR = 1544;   // mem bf16 row stride: 3088B (16B aligned), 772dw%64=4 -> conflict-free b128
static constexpr int HSTR = 72;     // small bf16 tile stride: 144B (16B aligned), 36dw%64=36 -> conflict-free
static constexpr int FSTR = 68;     // fp32 tile stride (68 % 64 = 4)

// workspace layout (bf16 elements): W^T, K contiguous per output column
static constexpr size_t WS_WK = 0;                        // [64][1536]
static constexpr size_t WS_WV = (size_t)Dsz * KTOT;       // [64][1536]
static constexpr size_t WS_WQ = 2 * (size_t)Dsz * KTOT;   // [64][64]
static constexpr size_t WS_WO = WS_WQ + Dsz * Dsz;        // [64][64]
static constexpr size_t WS_ELEMS = WS_WO + Dsz * Dsz;     // 204800 elems = 409600 B

union Frag16 { uint4 q[2]; v16bf v; };

DEV unsigned short f2bf(float f) {                 // fp32 -> bf16, round-to-nearest-even
    unsigned int u = __float_as_uint(f);
    unsigned int r = u + 0x7FFFu + ((u >> 16) & 1u);
    return (unsigned short)(r >> 16);
}

// A fragment (16x32 bf16, row-major in LDS). ISA layout:
// lane 0-15: M=lane; lane 16-31: M=lane-16, K offset +8.
// VGPR j<4: K = hi*8 + 2j,2j+1 ; VGPR 4..7: K = 16 + hi*8 + 2(j-4),+1.
// -> two contiguous 8-halfword runs: [k0+hi*8, +8) and [k0+16+hi*8, +8) => 2x ds_load_b128.
DEV v16bf load_a(const unsigned short* base, int stride, int k0, int lane) {
    const int row = lane & 15, hi = lane >> 4;
    const unsigned short* p = base + row * stride + k0 + hi * 8;
    Frag16 f;
    f.q[0] = *(const uint4*)(p);
    f.q[1] = *(const uint4*)(p + 16);
    return f.v;
}

// B fragment (32x16 bf16) from pre-transposed bf16 weights Wt[n][k] (ldk = K stride).
// lanes 0-15: N=n0+lane, K=k0..k0+7,k0+16..k0+23; lanes 16-31: N=n0+lane-16, K=k0+8..k0+15,k0+24..k0+31.
// same K order as load_a: two 8-halfword runs => 2x global_load_b128.
DEV v16bf load_bt(const unsigned short* __restrict__ Wt, int ldk, int k0, int n0, int lane) {
    const unsigned short* p = Wt + (size_t)(n0 + (lane & 15)) * ldk + k0 + (lane >> 4) * 8;
    Frag16 f;
    f.q[0] = *(const uint4*)(p);
    f.q[1] = *(const uint4*)(p + 16);
    return f.v;
}

DEV v8f wmma_bf16(v16bf a, v16bf b, v8f c) {
    return __builtin_amdgcn_wmma_f32_16x16x32_bf16(false, a, false, b, (short)0, c,
                                                   false, false);
}

// ---------------- prep: transpose + fp32->bf16 the four weight matrices ----------------
__global__ __launch_bounds__(256)
void prep_weights(const float* __restrict__ Wk, const float* __restrict__ Wv,
                  const float* __restrict__ Wq, const float* __restrict__ Wo,
                  unsigned short* __restrict__ ws)
{
    const int NK = Dsz * KTOT;             // 98304
    int idx = blockIdx.x * 256 + threadIdx.x;
    if (idx < NK) {                                       // Wk^T
        int n = idx / KTOT, k = idx % KTOT;
        ws[WS_WK + idx] = f2bf(Wk[(size_t)k * Dsz + n]);
    } else if (idx < 2 * NK) {                            // Wv^T
        int t = idx - NK;
        int n = t / KTOT, k = t % KTOT;
        ws[idx] = f2bf(Wv[(size_t)k * Dsz + n]);
    } else if (idx < 2 * NK + Dsz * Dsz) {                // Wq^T
        int t = idx - 2 * NK;
        int n = t >> 6, k = t & 63;
        ws[idx] = f2bf(Wq[k * Dsz + n]);
    } else if (idx < 2 * NK + 2 * Dsz * Dsz) {            // Wo^T
        int t = idx - 2 * NK - Dsz * Dsz;
        int n = t >> 6, k = t & 63;
        ws[idx] = f2bf(Wo[k * Dsz + n]);
    }
}

// ---------------- fused engram-memory kernel ----------------
__global__ __launch_bounds__(128)
void engram_fused(const long long* __restrict__ tok,
                  const float* __restrict__ hidden,
                  const float* __restrict__ tab2, const long long* __restrict__ m2,
                  const float* __restrict__ tab3, const long long* __restrict__ m3,
                  const float* __restrict__ tab4, const long long* __restrict__ m4,
                  const unsigned short* __restrict__ wbf,
                  const float* __restrict__ bq, const float* __restrict__ bk,
                  const float* __restrict__ bv, const float* __restrict__ bo,
                  float* __restrict__ out)
{
    extern __shared__ char smem[];
    long long*      s_tok  = (long long*)smem;                    // 24
    long long*      s_mult = s_tok + 24;                          // 24
    int*            s_idx  = (int*)(s_mult + 24);                 // 384
    float*          s_kt   = (float*)(s_idx + 384);               // 16*68
    float*          s_vt   = s_kt + TM * FSTR;                    // 16*68
    float*          s_qt   = s_vt + TM * FSTR;                    // 16*68
    float*          s_gate = s_qt + TM * FSTR;                    // 16
    unsigned short* s_mem  = (unsigned short*)(s_gate + 16);      // 16*1544 bf16 (16B-aligned)
    unsigned short* s_hid  = s_mem + TM * MSTR;                   // 16*72  bf16
    unsigned short* s_gv   = s_hid + TM * HSTR;                   // 16*72  bf16

    const unsigned short* wKt = wbf + WS_WK;
    const unsigned short* wVt = wbf + WS_WV;
    const unsigned short* wQt = wbf + WS_WQ;
    const unsigned short* wOt = wbf + WS_WO;

    const int tid   = threadIdx.x;
    const int lane  = tid & 31;
    const int wave  = tid >> 5;
    const int tiles = Ssz / TM;
    const int b     = blockIdx.x / tiles;
    const int s0    = (blockIdx.x % tiles) * TM;

    // ---- tokens (s0-3 .. s0+15) and per-head multipliers ----
    if (tid < 19) {
        int gs = s0 + tid - 3;
        s_tok[tid] = (gs >= 0) ? tok[(size_t)b * Ssz + gs] : 0ll;
    }
    if (tid < 24) {
        const long long* mp = (tid < 8) ? m2 : (tid < 16 ? m3 : m4);
        s_mult[tid] = mp[tid & 7];
    }
    __syncthreads();

    // ---- rolling n-gram hashes: 16 rows x 24 (ngram,head) segments ----
    for (int seg = tid; seg < TM * NSEG; seg += 128) {
        int r = seg / NSEG, c = seg % NSEG;
        int ng = c >> 3;
        int n  = ng + 2;
        int s  = s0 + r;
        int idx = -1;
        if (s >= n - 1) {
            long long P  = (ng == 0) ? 100003ll : ((ng == 1) ? 100019ll : 100043ll);
            long long mu = s_mult[c];
            long long hv = 0;
            for (int i = 0; i < n; ++i) {
                long long t = s_tok[r + 3 - (n - 1) + i];
                hv = (hv * mu + t) % P;
            }
            idx = (int)hv;
        }
        s_idx[seg] = idx;
    }
    __syncthreads();

    // ---- gather table rows -> bf16 mem tile in LDS (16 lanes per 64-float row) ----
    {
        const int q16 = tid & 15;
        for (int seg = tid >> 4; seg < TM * NSEG; seg += 8) {
            int r = seg / NSEG, c = seg % NSEG;
            int ng = c >> 3, h = c & 7;
            int idx = s_idx[seg];
            unsigned int pk0 = 0, pk1 = 0;
            if (idx >= 0) {
                const float* tab = (ng == 0) ? tab2 : ((ng == 1) ? tab3 : tab4);
                long long P = (ng == 0) ? 100003ll : ((ng == 1) ? 100019ll : 100043ll);
                const float4 val =
                    *(const float4*)(tab + ((size_t)h * (size_t)P + (size_t)idx) * Dsz + q16 * 4);
                pk0 = (unsigned)f2bf(val.x) | ((unsigned)f2bf(val.y) << 16);
                pk1 = (unsigned)f2bf(val.z) | ((unsigned)f2bf(val.w) << 16);
            }
            unsigned short* dst = s_mem + r * MSTR + c * Dsz + q16 * 4;
            *(unsigned int*)(dst)     = pk0;
            *(unsigned int*)(dst + 2) = pk1;
        }
    }
    // ---- hidden tile -> bf16 ----
    for (int t = tid; t < TM * 16; t += 128) {
        int r = t >> 4, q = t & 15;
        const float4 val = *(const float4*)(hidden + ((size_t)(b * Ssz + s0 + r)) * Dsz + q * 4);
        unsigned short* dst = s_hid + r * HSTR + q * 4;
        *(unsigned int*)(dst)     = (unsigned)f2bf(val.x) | ((unsigned)f2bf(val.y) << 16);
        *(unsigned int*)(dst + 2) = (unsigned)f2bf(val.z) | ((unsigned)f2bf(val.w) << 16);
    }
    __syncthreads();

    // ---- k / v projections: [16 x 1536] @ [1536 x 64], wave w owns cols 16w..16w+15 ----
    const int n0 = wave * 16;
    v8f acck = {}, accv = {};
    for (int k0 = 0; k0 < KTOT; k0 += 32) {
        v16bf a  = load_a(s_mem, MSTR, k0, lane);
        v16bf bK = load_bt(wKt, KTOT, k0, n0, lane);
        v16bf bV = load_bt(wVt, KTOT, k0, n0, lane);
        if (k0 + 32 < KTOT) {
            const size_t nxt = (size_t)(n0 + (lane & 15)) * KTOT + k0 + 32 + (lane >> 4) * 16;
            __builtin_prefetch(wKt + nxt, 0, 1);
            __builtin_prefetch(wVt + nxt, 0, 1);
        }
        acck = wmma_bf16(a, bK, acck);
        accv = wmma_bf16(a, bV, accv);
    }

    // ---- q projection: [16 x 64] @ [64 x 64] ----
    v8f accq = {};
    for (int k0 = 0; k0 < Dsz; k0 += 32) {
        v16bf a  = load_a(s_hid, HSTR, k0, lane);
        v16bf bQ = load_bt(wQt, Dsz, k0, n0, lane);
        accq = wmma_bf16(a, bQ, accq);
    }

    // ---- spill q/k/v tiles (+bias) to LDS. C/D layout: VGPR j -> M=j+8*hi, lane -> N ----
    {
        const int hi = lane >> 4;
        const int nc = n0 + (lane & 15);
        const float bkv = bk[nc], bvv = bv[nc], bqv = bq[nc];
#pragma unroll
        for (int j = 0; j < 8; ++j) {
            int M = j + hi * 8;
            s_kt[M * FSTR + nc] = acck[j] + bkv;
            s_vt[M * FSTR + nc] = accv[j] + bvv;
            s_qt[M * FSTR + nc] = accq[j] + bqv;
        }
    }
    __syncthreads();

    // ---- gate = sigmoid(q.k / sqrt(D)) ----
    if (tid < TM) {
        float s = 0.f;
        for (int c = 0; c < Dsz; ++c) s += s_qt[tid * FSTR + c] * s_kt[tid * FSTR + c];
        float g = 1.0f / (1.0f + __expf(-s * 0.125f));
        s_gate[tid] = g;
        out[(size_t)Bsz * Ssz * Dsz + (size_t)b * Ssz + s0 + tid] = g;
    }
    __syncthreads();

    // ---- gated v -> bf16 ----
    for (int t = tid; t < TM * Dsz; t += 128) {
        int r = t >> 6, c = t & 63;
        s_gv[r * HSTR + c] = f2bf(s_gate[r] * s_vt[r * FSTR + c]);
    }
    __syncthreads();

    // ---- output projection: (gate*v) @ Wo + bo ----
    v8f acco = {};
    for (int k0 = 0; k0 < Dsz; k0 += 32) {
        v16bf a  = load_a(s_gv, HSTR, k0, lane);
        v16bf bO = load_bt(wOt, Dsz, k0, n0, lane);
        acco = wmma_bf16(a, bO, acco);
    }
    {
        const int hi = lane >> 4;
        const int nc = n0 + (lane & 15);
        const float bov = bo[nc];
#pragma unroll
        for (int j = 0; j < 8; ++j) {
            int M = j + hi * 8;
            out[((size_t)b * Ssz + s0 + M) * Dsz + nc] = acco[j] + bov;
        }
    }
}

extern "C" void kernel_launch(void* const* d_in, const int* in_sizes, int n_in,
                              void* d_out, int out_size, void* d_ws, size_t ws_size,
                              hipStream_t stream) {
    (void)in_sizes; (void)n_in; (void)ws_size; (void)out_size;
    const long long* tok    = (const long long*)d_in[0];
    const float*     hidden = (const float*)d_in[1];
    const float*     tab2   = (const float*)d_in[2];
    const long long* m2     = (const long long*)d_in[3];
    const float*     tab3   = (const float*)d_in[4];
    const long long* m3     = (const long long*)d_in[5];
    const float*     tab4   = (const float*)d_in[6];
    const long long* m4     = (const long long*)d_in[7];
    const float*     Wq     = (const float*)d_in[8];
    const float*     bq     = (const float*)d_in[9];
    const float*     Wk     = (const float*)d_in[10];
    const float*     bk     = (const float*)d_in[11];
    const float*     Wv     = (const float*)d_in[12];
    const float*     bv     = (const float*)d_in[13];
    const float*     Wo     = (const float*)d_in[14];
    const float*     bo     = (const float*)d_in[15];
    float*           out    = (float*)d_out;
    unsigned short*  wbf    = (unsigned short*)d_ws;

    // 1) transpose + convert weights to fragment-ready bf16 in workspace
    {
        int total = (int)WS_ELEMS;                 // 204800
        dim3 grid((total + 255) / 256);
        prep_weights<<<grid, 256, 0, stream>>>(Wk, Wv, Wq, Wo, wbf);
    }

    // 2) fused gather + projections + gate + output
    constexpr size_t SMEM =
        24 * 8 + 24 * 8 + 384 * 4 +                 // tokens, mults, hash indices
        3 * (size_t)TM * FSTR * 4 + 16 * 4 +        // q/k/v fp32 tiles, gate
        (size_t)TM * MSTR * 2 +                     // mem bf16 tile
        (size_t)TM * HSTR * 2 +                     // hidden bf16 tile
        (size_t)TM * HSTR * 2;                      // gated-v bf16 tile

    dim3 grid(Bsz * (Ssz / TM));
    engram_fused<<<grid, 128, SMEM, stream>>>(tok, hidden, tab2, m2, tab3, m3, tab4, m4,
                                              wbf, bq, bk, bv, bo, out);
}
